// Net_ASAP_11227044511910
// MI455X (gfx1250) — matrix operand order repaired and checked by reading, not compiled, not run on hardware
//
#include <hip/hip_runtime.h>
#include <hip/hip_bf16.h>

// ---------------------------------------------------------------------------
// CDNA5 (gfx1250) fused ASAP-GNN forward.
// One workgroup (8 wave32) per graph; all matmuls via V_WMMA_F32_16X16X32_BF16.
// All global accesses go through explicit address_space(1) pointers so they
// lower to global_load/global_store (LOADcnt/STOREcnt only) instead of
// flat_load (which also ticks DScnt and entangles with the LDS WMMA-fragment
// s_wait_dscnt chain; ISA 10.1.1).
// ---------------------------------------------------------------------------

typedef const float __attribute__((address_space(1)))* gcf;
typedef float       __attribute__((address_space(1)))* gf;

typedef __attribute__((ext_vector_type(16))) __bf16 v16b;
typedef __attribute__((ext_vector_type(8)))  float  v8f;

__device__ __forceinline__ v8f wmma_bf(v16b a, v16b b, v8f c) {
  // (neg_a, A, neg_b, B, c_mod, C, reuse_a, reuse_b)
  return __builtin_amdgcn_wmma_f32_16x16x32_bf16(false, a, false, b, (short)0, c,
                                                 false, false);
}

// A fragment: 16x32 bf16 tile, row-major in LDS with leading dim ld.
// lanes 0-15 -> M=lane, K = 0..15; lanes 16-31 -> K = 16..31.
__device__ __forceinline__ v16b ldsA(const __bf16* base, int ld, int lane) {
  int row = lane & 15;
  int kb  = (lane >> 4) << 4; // 0 or 16
  const __bf16* p = base + row * ld;
  v16b a;
#pragma unroll
  for (int i = 0; i < 16; ++i) {
    a[i] = p[kb + i];
  }
  return a;
}

// B fragment: 32x16 bf16, source row-major [K x N] in LDS, leading dim ld.
// lanes 0-15 -> N=lane, K=0..15; lanes 16-31 -> K=16..31.
__device__ __forceinline__ v16b ldsB_rm(const __bf16* base, int ld, int lane) {
  int n  = lane & 15;
  int kb = (lane >> 4) << 4; // 0 or 16
  v16b b;
#pragma unroll
  for (int i = 0; i < 8; ++i) {
    b[2 * i]     = base[(kb + 2 * i) * ld + n];
    b[2 * i + 1] = base[(kb + 2 * i + 1) * ld + n];
  }
  return b;
}

// B fragment from f32 weight W[HOUT x HIN] (row-major, global/L2):
// computing X @ W^T so B[k][n] = W[n0+n][k0+k]; contiguous in k.
__device__ __forceinline__ v16b gW_B(gcf W, int ld, int n0, int k0, int lane) {
  int n  = lane & 15;
  int kb = (lane >> 4) << 4;
  gcf p = W + (size_t)(n0 + n) * ld + k0 + kb;
  v16b b;
#pragma unroll
  for (int i = 0; i < 16; ++i) b[i] = (__bf16)p[i];
  return b;
}

// D store: 16x16 f32 tile to LDS, leading dim ld.
__device__ __forceinline__ void stD(float* dst, int ld, v8f d, int lane) {
  int n  = lane & 15;
  int mb = (lane >> 4) << 3;
#pragma unroll
  for (int r = 0; r < 8; ++r) dst[(mb + r) * ld + n] = d[r];
}

// LDS row . global weight vector
__device__ __forceinline__ float dot128(const float* a, gcf b) {
  float s = 0.f;
#pragma unroll 8
  for (int q = 0; q < 128; ++q) s += a[q] * b[q];
  return s;
}

// Warm a 128x128 f32 weight matrix (64 KB = 512 cachelines) into cache.
// Lowers to global_prefetch_b8 on gfx1250; no LOADcnt cost.
__device__ __forceinline__ void prefetch_w(const float* W, int tid) {
  for (int e = tid; e < 512; e += 256)
    __builtin_prefetch(W + e * 32, 0, 1);
}

// ------------------------- graph conv (relu fused) -------------------------
__device__ void run_conv(int tid, int nv, float* h, float* adjL, __bf16* hb,
                         __bf16* gb, __bf16* ab, float* rcnt, const float* wr_,
                         const float* br_, const float* wroot_) {
  const int lane = tid & 31, wave = tid >> 5;
  gcf wr = (gcf)wr_, br = (gcf)br_, wroot = (gcf)wroot_;
  prefetch_w(wr_, tid);
  prefetch_w(wroot_, tid);
  for (int e = tid; e < 64 * 128; e += 256) hb[e] = (__bf16)h[e];
  for (int e = tid; e < 64 * 64; e += 256) ab[e] = (__bf16)adjL[e];
  if (tid < 64) {
    float c = 0.f;
    if (tid < nv)
      for (int j = 0; j < 64; ++j) c += (adjL[tid * 64 + j] != 0.f) ? 1.f : 0.f;
    rcnt[tid] = 1.f / fmaxf(c, 1.f); // reciprocal: mul instead of div below
  }
  __syncthreads();
  const int rt = (nv + 15) >> 4;
  // agg = (adj @ h) * rcnt  -> gb (bf16)
  for (int t = wave; t < rt * 8; t += 8) {
    int tr = t >> 3, tc = t & 7;
    v8f acc = {};
    for (int k0 = 0; k0 < 64; k0 += 32) {
      v16b a = ldsA(ab + tr * 16 * 64 + k0, 64, lane);
      v16b b = ldsB_rm(hb + k0 * 128 + tc * 16, 128, lane);
      acc = wmma_bf(a, b, acc);
    }
    int n = lane & 15, mb = (lane >> 4) << 3;
#pragma unroll
    for (int r = 0; r < 8; ++r) {
      int rr = tr * 16 + mb + r;
      gb[rr * 128 + tc * 16 + n] = (__bf16)(acc[r] * rcnt[rr]);
    }
  }
  __syncthreads();
  // h = relu(x @ wroot^T + agg @ wr^T + br); zero padded rows
  for (int t = wave; t < rt * 8; t += 8) {
    int tr = t >> 3, tc = t & 7;
    v8f acc = {};
    for (int k0 = 0; k0 < 128; k0 += 32) {
      v16b a = ldsA(hb + tr * 16 * 128 + k0, 128, lane);
      v16b b = gW_B(wroot, 128, tc * 16, k0, lane);
      acc = wmma_bf(a, b, acc);
    }
    for (int k0 = 0; k0 < 128; k0 += 32) {
      v16b a = ldsA(gb + tr * 16 * 128 + k0, 128, lane);
      v16b b = gW_B(wr, 128, tc * 16, k0, lane);
      acc = wmma_bf(a, b, acc);
    }
    int n = lane & 15, mb = (lane >> 4) << 3;
    float bias = br[tc * 16 + n];
#pragma unroll
    for (int r = 0; r < 8; ++r) {
      int rr = tr * 16 + mb + r;
      h[rr * 128 + tc * 16 + n] = (rr < nv) ? fmaxf(acc[r] + bias, 0.f) : 0.f;
    }
  }
  __syncthreads();
}

// ------------------------------ ASAP pooling -------------------------------
__device__ void run_pool(int tid, int n, int k, float* h, float* adjL,
                         __bf16* hb, __bf16* ab, __bf16* gb, float* tmp,
                         float* sbuf, float* s1v, float* s2v, float* av,
                         float* bv, float* c3v, float* degv, float* fitv,
                         float* vals, int* perm, float* Mbuf, float* adjn_s,
                         const float* lw_, const float* lb_, const float* aw_,
                         float attb, const float* w1_, float b1,
                         const float* w2_, const float* w3_, float b3) {
  const int lane = tid & 31, wave = tid >> 5;
  const int rt = (n + 15) >> 4;
  gcf lw = (gcf)lw_, lb = (gcf)lb_, aw = (gcf)aw_;
  gcf w1 = (gcf)w1_, w2 = (gcf)w2_, w3 = (gcf)w3_;
  prefetch_w(lw_, tid);
  if (tid < n) { // adj1 = adj with diag forced nonzero
    float d = adjL[tid * 64 + tid];
    adjL[tid * 64 + tid] = (d == 0.f) ? 1.f : d;
  }
  for (int e = tid; e < 64 * 128; e += 256) hb[e] = (__bf16)h[e];
  __syncthreads();
  // xq: masked neighborhood max -> gb (bf16)
  for (int idx = tid; idx < n * 128; idx += 256) {
    int i = idx >> 7, f = idx & 127;
    float m = -1e30f;
    for (int j = 0; j < n; ++j)
      if (adjL[i * 64 + j] != 0.f) m = fmaxf(m, h[j * 128 + f]);
    gb[i * 128 + f] = (__bf16)m;
  }
  for (int idx = n * 128 + tid; idx < rt * 16 * 128; idx += 256)
    gb[idx] = (__bf16)0.f;
  __syncthreads();
  // xq2 = xq @ lw^T + lb -> tmp (f32)
  for (int t = wave; t < rt * 8; t += 8) {
    int tr = t >> 3, tc = t & 7;
    v8f acc = {};
    for (int k0 = 0; k0 < 128; k0 += 32) {
      v16b a = ldsA(gb + tr * 16 * 128 + k0, 128, lane);
      v16b b = gW_B(lw, 128, tc * 16, k0, lane);
      acc = wmma_bf(a, b, acc);
    }
    int nn = lane & 15, mb = (lane >> 4) << 3;
    float bias = lb[tc * 16 + nn];
#pragma unroll
    for (int r = 0; r < 8; ++r)
      tmp[(tr * 16 + mb + r) * 128 + tc * 16 + nn] = acc[r] + bias;
  }
  __syncthreads();
  if (tid < 64) {
    if (tid < n) s1v[tid] = dot128(tmp + tid * 128, aw);
  } else if (tid < 128) {
    int i = tid - 64;
    if (i < n) s2v[i] = dot128(h + i * 128, aw + 128);
  }
  __syncthreads();
  // masked leaky-relu softmax rows -> sbuf
  if (tid < n) {
    int i = tid;
    float mx = -1e30f;
    for (int j = 0; j < n; ++j)
      if (adjL[i * 64 + j] != 0.f) {
        float e = s1v[i] + s2v[j] + attb;
        e = (e > 0.f) ? e : 0.2f * e;
        mx = fmaxf(mx, e);
      }
    float sum = 0.f;
    for (int j = 0; j < n; ++j) {
      float v = 0.f;
      if (adjL[i * 64 + j] != 0.f) {
        float e = s1v[i] + s2v[j] + attb;
        e = (e > 0.f) ? e : 0.2f * e;
        v = __expf(e - mx);
        sum += v;
      }
      sbuf[i * 64 + j] = v;
    }
    float inv = 1.f / sum;
    for (int j = 0; j < 64; ++j)
      sbuf[i * 64 + j] = (j < n) ? sbuf[i * 64 + j] * inv : 0.f;
  }
  __syncthreads();
  for (int idx = tid; idx < rt * 16 * 64; idx += 256) {
    int i = idx >> 6;
    ab[idx] = (i < n) ? (__bf16)sbuf[idx] : (__bf16)0.f;
  }
  __syncthreads();
  // xc = s @ h -> tmp
  for (int t = wave; t < rt * 8; t += 8) {
    int tr = t >> 3, tc = t & 7;
    v8f acc = {};
    for (int k0 = 0; k0 < 64; k0 += 32) {
      v16b a = ldsA(ab + tr * 16 * 64 + k0, 64, lane);
      v16b b = ldsB_rm(hb + k0 * 128 + tc * 16, 128, lane);
      acc = wmma_bf(a, b, acc);
    }
    stD(tmp + tr * 16 * 128 + tc * 16, 128, acc, lane);
  }
  __syncthreads();
  if (tid < n) {
    av[tid]  = dot128(tmp + tid * 128, w1) + b1;
    bv[tid]  = dot128(tmp + tid * 128, w2);
    c3v[tid] = dot128(tmp + tid * 128, w3);
    float dg = 0.f;
    for (int j = 0; j < 64; ++j) dg += adjL[tid * 64 + j];
    degv[tid] = dg;
  }
  __syncthreads();
  if (tid < n) {
    float t0 = 0.f;
    for (int j = 0; j < n; ++j) t0 += adjL[tid * 64 + j] * av[j];
    float z = t0 - bv[tid] * degv[tid] + c3v[tid] + b3;
    fitv[tid] = 1.f / (1.f + __expf(-z));
  }
  __syncthreads();
  if (tid == 0) { // top-k (k <= 16, n <= 64): serial selection
    for (int t = 0; t < k; ++t) {
      float best = -1e30f;
      int bi = 0;
      for (int i = 0; i < n; ++i)
        if (fitv[i] > best) { best = fitv[i]; bi = i; }
      vals[t] = best; perm[t] = bi; fitv[bi] = -1e30f;
    }
  }
  __syncthreads();
  // h' = gather(xc, perm) * vals ; zero padding rows
  for (int idx = tid; idx < 64 * 128; idx += 256) {
    int t = idx >> 7, f = idx & 127;
    h[idx] = (t < k) ? tmp[perm[t] * 128 + f] * vals[t] : 0.f;
  }
  // M = sc @ adj1
  for (int idx = tid; idx < k * 64; idx += 256) {
    int t = idx >> 6, j2 = idx & 63;
    float m = 0.f;
    const float* sr = sbuf + perm[t] * 64;
    for (int j = 0; j < n; ++j) m += sr[j] * adjL[j * 64 + j2];
    Mbuf[idx] = m;
  }
  __syncthreads();
  // adjn = M @ sc^T, zero diagonal
  for (int idx = tid; idx < k * k; idx += 256) {
    int t = idx / k, u = idx % k;
    float v = 0.f;
    const float* su = sbuf + perm[u] * 64;
    for (int j2 = 0; j2 < n; ++j2) v += Mbuf[t * 64 + j2] * su[j2];
    adjn_s[t * 16 + u] = (t == u) ? 0.f : v;
  }
  __syncthreads();
  for (int idx = tid; idx < 64 * 64; idx += 256) adjL[idx] = 0.f;
  __syncthreads();
  for (int idx = tid; idx < k * k; idx += 256)
    adjL[(idx / k) * 64 + (idx % k)] = adjn_s[(idx / k) * 16 + (idx % k)];
  __syncthreads();
}

__device__ void row_mean(int tid, int nv, const float* h, float* zdst) {
  if (tid < 128) {
    float s = 0.f;
    for (int i = 0; i < nv; ++i) s += h[i * 128 + tid];
    zdst[tid] = s / (float)nv;
  }
  __syncthreads();
}

// --------------------------------- kernel ----------------------------------
__global__ __launch_bounds__(256, 1) void asap_net_kernel(
    const float* __restrict__ x, const float* __restrict__ adj_g,
    const float* c1_wr, const float* c1_br, const float* c1_wroot,
    const float* cs_wr, const float* cs_br, const float* cs_wroot,
    const float* pl_w, const float* pl_b, const float* pa_w, const float* pa_b,
    const float* p1_w, const float* p1_b, const float* p2_w, const float* p3_w,
    const float* p3_b, const float* l1_w, const float* l1_b, const float* l2_w,
    const float* l2_b, float* __restrict__ out) {
  __shared__ float  h[64 * 128];
  __shared__ float  tmp[64 * 128];
  __shared__ __bf16 hb[64 * 128];
  __shared__ __bf16 gb[64 * 128];
  __shared__ __bf16 ab[64 * 64];
  __shared__ float  adjL[64 * 64];
  __shared__ float  sbuf[64 * 64];
  __shared__ float  rcnt[64], s1v[64], s2v[64], av[64], bv[64], c3v[64],
      degv[64], fitv[64];
  __shared__ float  vals[16];
  __shared__ int    perm[16];
  __shared__ float  Mbuf[16 * 64];
  __shared__ float  adjn_s[16 * 16];
  __shared__ float  zbuf[5 * 128];
  __shared__ float  z1[128];
  __shared__ float  osm[10];

  const int g = blockIdx.x;
  const int tid = threadIdx.x;

  {
    gcf xg = (gcf)x;
    gcf ag = (gcf)adj_g;
    for (int e = tid; e < 64 * 128; e += 256) h[e] = xg[(size_t)g * 8192 + e];
    for (int e = tid; e < 64 * 64; e += 256)
      adjL[e] = ag[(size_t)g * 4096 + e];
  }
  __syncthreads();

  int nv = 64;
  run_conv(tid, nv, h, adjL, hb, gb, ab, rcnt, c1_wr, c1_br, c1_wroot);
  row_mean(tid, nv, h, zbuf);

  for (int i = 0; i < 4; ++i) {
    run_conv(tid, nv, h, adjL, hb, gb, ab, rcnt, cs_wr + (size_t)i * 128 * 128,
             cs_br + i * 128, cs_wroot + (size_t)i * 128 * 128);
    row_mean(tid, nv, h, zbuf + (i + 1) * 128);
    if (i == 0 || i == 2) {
      int p = i >> 1;
      int kk = (p == 0) ? 16 : 4;
      run_pool(tid, nv, kk, h, adjL, hb, ab, gb, tmp, sbuf, s1v, s2v, av, bv,
               c3v, degv, fitv, vals, perm, Mbuf, adjn_s,
               pl_w + (size_t)p * 128 * 128, pl_b + p * 128, pa_w + p * 256,
               ((gcf)pa_b)[p], p1_w + p * 128, ((gcf)p1_b)[p], p2_w + p * 128,
               p3_w + p * 128, ((gcf)p3_b)[p]);
      nv = kk;
    }
  }
  __syncthreads();

  // lin1: 640 -> 128, relu
  if (tid < 128) {
    float acc = ((gcf)l1_b)[tid];
    gcf wrow = (gcf)l1_w + (size_t)tid * 640;
    for (int q = 0; q < 640; ++q) acc += zbuf[q] * wrow[q];
    z1[tid] = fmaxf(acc, 0.f);
  }
  __syncthreads();
  // lin2: 128 -> 10
  if (tid < 10)
    osm[tid] = ((gcf)l2_b)[tid] + dot128(z1, (gcf)l2_w + tid * 128);
  __syncthreads();
  if (tid == 0) {
    float m = osm[0];
    for (int c = 1; c < 10; ++c) m = fmaxf(m, osm[c]);
    float s = 0.f;
    for (int c = 0; c < 10; ++c) s += __expf(osm[c] - m);
    float ls = __logf(s);
    gf og = (gf)out;
    for (int c = 0; c < 10; ++c) og[(size_t)g * 10 + c] = osm[c] - m - ls;
  }
}

extern "C" void kernel_launch(void* const* d_in, const int* in_sizes, int n_in,
                              void* d_out, int out_size, void* d_ws,
                              size_t ws_size, hipStream_t stream) {
  (void)n_in; (void)out_size; (void)d_ws; (void)ws_size;
  const int G = in_sizes[0] / (64 * 128);
  asap_net_kernel<<<G, 256, 0, stream>>>(
      (const float*)d_in[0], (const float*)d_in[1], (const float*)d_in[2],
      (const float*)d_in[3], (const float*)d_in[4], (const float*)d_in[5],
      (const float*)d_in[6], (const float*)d_in[7], (const float*)d_in[8],
      (const float*)d_in[9], (const float*)d_in[10], (const float*)d_in[11],
      (const float*)d_in[12], (const float*)d_in[13], (const float*)d_in[14],
      (const float*)d_in[15], (const float*)d_in[16], (const float*)d_in[17],
      (const float*)d_in[18], (const float*)d_in[19], (const float*)d_in[20],
      (float*)d_out);
}
